// DeepSpeedSelfAttention_49752901157487
// MI455X (gfx1250) — matrix operand order repaired and checked by reading, not compiled, not run on hardware
//
#include <hip/hip_runtime.h>
#include <hip/hip_bf16.h>

typedef _Float16 v16h __attribute__((ext_vector_type(16)));
typedef _Float16 v8h  __attribute__((ext_vector_type(8)));
typedef _Float16 v4h  __attribute__((ext_vector_type(4)));
typedef float    v8f  __attribute__((ext_vector_type(8)));

#define B_   2
#define S_   2048
#define H_   1024
#define NH_  16
#define HD_  64
#define M_   (B_ * S_)   // 4096 tokens
#define N3_  (3 * H_)    // 3072
#define LDK  40          // LDS K-stride (32 + 8 pad halves), keeps 16B alignment

union V16 { v16h v; v8h h[2]; };

static __device__ __forceinline__ v8f wmma_f16(v16h a, v16h b, v8f c) {
  // D = A(16x32 f16) * B(32x16 f16) + C(16x16 f32)
  return __builtin_amdgcn_wmma_f32_16x16x32_f16(false, a, false, b, (short)0, c, false, false);
}

// CDNA5 async DMA: global -> LDS, 16B per lane, tracked by ASYNCcnt.
// VDST VGPR = per-lane LDS byte address (low 32 bits of generic LDS pointer,
// ISA 10.2: LDS aperture uses addr[31:0]).
static __device__ __forceinline__ void async_copy16(void* lds_dst, const void* gsrc) {
  unsigned lds_off = (unsigned)(size_t)lds_dst;
  asm volatile("global_load_async_to_lds_b128 %0, %1, off"
               :: "v"(lds_off), "v"(gsrc) : "memory");
}
static __device__ __forceinline__ void async_wait0() {
  asm volatile("s_wait_asynccnt 0x0" ::: "memory");
}
static __device__ __forceinline__ void lds_wait0() {
  asm volatile("s_wait_dscnt 0x0" ::: "memory");
}

// ---------------------------------------------------------------------------
// Kernel 1: LayerNorm(x) -> f16 xn [M_][H_]
// ---------------------------------------------------------------------------
__global__ __launch_bounds__(256) void k_ln(const float* __restrict__ x,
                                            const float* __restrict__ w,
                                            const float* __restrict__ b,
                                            _Float16* __restrict__ xn) {
  __shared__ float r1[256];
  __shared__ float r2[256];
  const int row = blockIdx.x, t = threadIdx.x;
  float4 v = ((const float4*)(x + (size_t)row * H_))[t];
  r1[t] = v.x + v.y + v.z + v.w;
  r2[t] = v.x * v.x + v.y * v.y + v.z * v.z + v.w * v.w;
  __syncthreads();
  for (int o = 128; o > 0; o >>= 1) {
    if (t < o) { r1[t] += r1[t + o]; r2[t] += r2[t + o]; }
    __syncthreads();
  }
  const float mu  = r1[0] * (1.0f / H_);
  const float var = r2[0] * (1.0f / H_) - mu * mu;
  const float inv = rsqrtf(var + 1e-5f);
  float4 wv = ((const float4*)w)[t];
  float4 bv = ((const float4*)b)[t];
  v4h o;
  o[0] = (_Float16)((v.x - mu) * inv * wv.x + bv.x);
  o[1] = (_Float16)((v.y - mu) * inv * wv.y + bv.y);
  o[2] = (_Float16)((v.z - mu) * inv * wv.z + bv.z);
  o[3] = (_Float16)((v.w - mu) * inv * wv.w + bv.w);
  *(v4h*)(xn + (size_t)row * H_ + t * 4) = o;
}

// ---------------------------------------------------------------------------
// Kernel 2: QKV GEMM + bias + rotary + scatter.
//   xn[4096x1024] (f16) @ W[1024x3072] (f32->f16) -> Q/K rotary -> f16 scatter
//   Qh,Kh: [BH][S][HD];  Vt: [BH][HD][S] (pre-transposed for PV B-fragments)
// ---------------------------------------------------------------------------
__global__ __launch_bounds__(256) void k_qkv(const _Float16* __restrict__ xn,
                                             const float* __restrict__ W,
                                             const float* __restrict__ bias,
                                             _Float16* __restrict__ Qh,
                                             _Float16* __restrict__ Kh,
                                             _Float16* __restrict__ Vt) {
  __shared__ __align__(16) char smem[34816];   // max(staging 15360, V-epilogue 34816)
  _Float16* ldsA  = (_Float16*)smem;                    // [128][LDK]
  _Float16* ldsBt = (_Float16*)(smem + 128 * LDK * 2);  // [64][LDK] (N-major)
  const int tid = threadIdx.x, wv = tid >> 5, l = tid & 31, lo = l & 15, hi = l >> 4;
  const int n0 = blockIdx.x * 64, m0 = blockIdx.y * 128;

  v8f acc[4];
  for (int f = 0; f < 4; ++f) for (int i = 0; i < 8; ++i) acc[f][i] = 0.f;

  for (int k0 = 0; k0 < H_; k0 += 32) {
    __syncthreads();
    // Stage A: 128 x 32 halves, async DMA (no VGPR round trip, ASYNCcnt)
#pragma unroll
    for (int it = 0; it < 2; ++it) {
      const int i = tid + it * 256;
      const int r = i >> 2, c = (i & 3) * 8;
      async_copy16(&ldsA[r * LDK + c], &xn[(size_t)(m0 + r) * H_ + k0 + c]);
    }
    // Stage B transposed: 32 x 64 f32 -> ldsBt[n][k] f16
#pragma unroll
    for (int it = 0; it < 2; ++it) {
      const int i = tid + it * 256;
      const int kk = i >> 4, c = (i & 15) * 4;
      float4 w4 = *(const float4*)&W[(size_t)(k0 + kk) * N3_ + n0 + c];
      if (k0 + 32 < H_)  // warm WGP$ for next chunk while this one computes
        __builtin_prefetch(&W[(size_t)(k0 + 32 + kk) * N3_ + n0 + c], 0, 1);
      ldsBt[(c + 0) * LDK + kk] = (_Float16)w4.x;
      ldsBt[(c + 1) * LDK + kk] = (_Float16)w4.y;
      ldsBt[(c + 2) * LDK + kk] = (_Float16)w4.z;
      ldsBt[(c + 3) * LDK + kk] = (_Float16)w4.w;
    }
    async_wait0();
    __syncthreads();
    V16 A;
    const _Float16* ap = &ldsA[(wv * 16 + lo) * LDK + hi * 8];
    A.h[0] = *(const v8h*)ap; A.h[1] = *(const v8h*)(ap + 16);
    V16 Bf[4];
#pragma unroll
    for (int f = 0; f < 4; ++f) {
      const _Float16* bp = &ldsBt[(f * 16 + lo) * LDK + hi * 8];
      Bf[f].h[0] = *(const v8h*)bp; Bf[f].h[1] = *(const v8h*)(bp + 16);
    }
#pragma unroll
    for (int f = 0; f < 4; ++f) acc[f] = wmma_f16(A.v, Bf[f].v, acc[f]);
  }

  // bias (per output column)
  for (int f = 0; f < 4; ++f) {
    const float bb = bias[n0 + f * 16 + lo];
    for (int r = 0; r < 8; ++r) acc[f][r] += bb;
  }

  const int sel   = n0 >> 10;              // 0=Q 1=K 2=V (n0 is 64-aligned, heads don't split)
  const int hh    = (n0 & (H_ - 1)) >> 6;  // head
  const int mrow0 = m0 + wv * 16;

  if (sel < 2) {
    // Rotary: rotary_dim == HD == 64; pairs (d, d+32) live in acc[f] / acc[f+2].
    const float qs = (sel == 0) ? 0.125f : 1.0f;  // fold 1/sqrt(64) into Q
    const float e0 = __expf((float)lo        * (-9.210340372f / 32.f)); // 10000^(-i/32)
    const float e1 = __expf((float)(lo + 16) * (-9.210340372f / 32.f));
    for (int r = 0; r < 8; ++r) {
      const int m = mrow0 + hi * 8 + r;
      const float s = (float)(m & (S_ - 1));
      float s0, c0, s1, c1;
      __sincosf(s * e0, &s0, &c0);
      __sincosf(s * e1, &s1, &c1);
      const float v0 = acc[0][r], v1 = acc[1][r], v2 = acc[2][r], v3 = acc[3][r];
      acc[0][r] = (v0 * c0 - v2 * s0) * qs;
      acc[2][r] = (v2 * c0 + v0 * s0) * qs;
      acc[1][r] = (v1 * c1 - v3 * s1) * qs;
      acc[3][r] = (v3 * c1 + v1 * s1) * qs;
    }
    _Float16* dst = (sel == 0) ? Qh : Kh;
    for (int f = 0; f < 4; ++f)
      for (int r = 0; r < 8; ++r) {
        const int m = mrow0 + hi * 8 + r;
        const int bb = m >> 11, s = m & (S_ - 1);
        dst[((size_t)(bb * NH_ + hh) * S_ + s) * HD_ + f * 16 + lo] = (_Float16)acc[f][r];
      }
  } else {
    // V: transpose 16x64 tile through LDS, store Vt[bh][d][s] (s-contiguous)
    __syncthreads();  // staging LDS no longer needed by any wave
    float* E = (float*)smem + wv * (16 * 68);
    for (int f = 0; f < 4; ++f)
      for (int r = 0; r < 8; ++r)
        E[(hi * 8 + r) * 68 + f * 16 + lo] = acc[f][r];
    lds_wait0();
    const int bb = mrow0 >> 11, s0r = mrow0 & (S_ - 1);
    for (int dd = l; dd < 64; dd += 32) {
      v16h o;
      for (int r = 0; r < 16; ++r) o[r] = (_Float16)E[r * 68 + dd];
      *(v16h*)&Vt[((size_t)(bb * NH_ + hh) * HD_ + dd) * S_ + s0r] = o;
    }
  }
}

// ---------------------------------------------------------------------------
// Kernel 3: causal flash attention. One wave per 16-row q-tile, 32-wide k-tiles.
//   scores WMMA (QK^T), online softmax, probs->LDS->A-frag, PV WMMA.
// ---------------------------------------------------------------------------
__global__ __launch_bounds__(128) void k_attn(const _Float16* __restrict__ Qh,
                                              const _Float16* __restrict__ Kh,
                                              const _Float16* __restrict__ Vt,
                                              const int* __restrict__ imask,
                                              _Float16* __restrict__ ctx) {
  __shared__ __align__(16) _Float16 ldsP[4][16 * LDK];
  const int tid = threadIdx.x, wv = tid >> 5, l = tid & 31, lo = l & 15, hi = l >> 4;
  const int wid = blockIdx.x * 4 + wv;
  const int bh = wid >> 7, qt = wid & 127;   // 128 q-tiles per (b,h)
  const int q0 = qt * 16;
  const int bb = bh >> 4, hh = bh & (NH_ - 1);
  const _Float16* Qb = Qh + (size_t)bh * S_ * HD_;
  const _Float16* Kb = Kh + (size_t)bh * S_ * HD_;
  const _Float16* Vb = Vt + (size_t)bh * HD_ * S_;

  V16 qa[2];  // A-fragments of Q, d-chunks [0,32) and [32,64)
  for (int dc = 0; dc < 2; ++dc) {
    const _Float16* qp = &Qb[(size_t)(q0 + lo) * HD_ + dc * 32 + hi * 8];
    qa[dc].h[0] = *(const v8h*)qp; qa[dc].h[1] = *(const v8h*)(qp + 16);
  }

  v8f cacc[4];
  for (int f = 0; f < 4; ++f) for (int i = 0; i < 8; ++i) cacc[f][i] = 0.f;
  float mrun[8], lrun[8];
  for (int r = 0; r < 8; ++r) { mrun[r] = -3.0e38f; lrun[r] = 0.f; }

  _Float16* P = &ldsP[wv][0];
  const int ntiles = (q0 + 16 + 31) >> 5;  // causal: only tiles with k0 <= q0+15
  for (int kt = 0; kt < ntiles; ++kt) {
    const int k0 = kt * 32;
    // Load all K B-fragments first, then issue the score WMMAs.
    V16 kb[2][2];
#pragma unroll
    for (int c = 0; c < 2; ++c) {
      const _Float16* kp = &Kb[(size_t)(k0 + c * 16 + lo) * HD_ + hi * 8];
      kb[c][0].h[0] = *(const v8h*)kp;        kb[c][0].h[1] = *(const v8h*)(kp + 16);
      kb[c][1].h[0] = *(const v8h*)(kp + 32); kb[c][1].h[1] = *(const v8h*)(kp + 48);
    }
    v8f sc[2];
#pragma unroll
    for (int c = 0; c < 2; ++c) {
      v8f z; for (int i = 0; i < 8; ++i) z[i] = 0.f;
      z     = wmma_f16(qa[0].v, kb[c][0].v, z);
      sc[c] = wmma_f16(qa[1].v, kb[c][1].v, z);
    }
    const float mb0 = (1.f - (float)imask[bb * S_ + k0 + lo])      * -10000.f;
    const float mb1 = (1.f - (float)imask[bb * S_ + k0 + 16 + lo]) * -10000.f;
    for (int r = 0; r < 8; ++r) {
      const int q = q0 + hi * 8 + r;
      float s0v = (k0 + lo      <= q) ? sc[0][r] + mb0 : -10000.f;
      float s1v = (k0 + 16 + lo <= q) ? sc[1][r] + mb1 : -10000.f;
      float t = fmaxf(s0v, s1v);
      t = fmaxf(t, __shfl_xor(t, 1, 32));
      t = fmaxf(t, __shfl_xor(t, 2, 32));
      t = fmaxf(t, __shfl_xor(t, 4, 32));
      t = fmaxf(t, __shfl_xor(t, 8, 32));
      const float mnew  = fmaxf(mrun[r], t);
      const float alpha = __expf(mrun[r] - mnew);
      const float p0 = __expf(s0v - mnew);
      const float p1 = __expf(s1v - mnew);
      float ps = p0 + p1;
      ps += __shfl_xor(ps, 1, 32);
      ps += __shfl_xor(ps, 2, 32);
      ps += __shfl_xor(ps, 4, 32);
      ps += __shfl_xor(ps, 8, 32);
      lrun[r] = lrun[r] * alpha + ps;
      mrun[r] = mnew;
      for (int f = 0; f < 4; ++f) cacc[f][r] *= alpha;
      P[(hi * 8 + r) * LDK + lo]      = (_Float16)p0;
      P[(hi * 8 + r) * LDK + 16 + lo] = (_Float16)p1;
    }
    lds_wait0();  // C-layout -> A-layout handoff (same-wave DS ordering)
    V16 pa;
    const _Float16* pp = &P[lo * LDK + hi * 8];
    pa.h[0] = *(const v8h*)pp; pa.h[1] = *(const v8h*)(pp + 16);
    // Load all V B-fragments (pre-transposed Vt: contiguous k halves), then WMMAs.
    V16 vb[4];
#pragma unroll
    for (int f = 0; f < 4; ++f) {
      const _Float16* vp = &Vb[(size_t)(f * 16 + lo) * S_ + k0 + hi * 8];
      vb[f].h[0] = *(const v8h*)vp; vb[f].h[1] = *(const v8h*)(vp + 16);
    }
#pragma unroll
    for (int f = 0; f < 4; ++f) cacc[f] = wmma_f16(pa.v, vb[f].v, cacc[f]);
  }
  for (int r = 0; r < 8; ++r) {
    const int q = q0 + hi * 8 + r;
    const float inv = 1.f / lrun[r];
    for (int f = 0; f < 4; ++f)
      ctx[((size_t)(bb * S_ + q)) * H_ + hh * HD_ + f * 16 + lo] = (_Float16)(cacc[f][r] * inv);
  }
}

// ---------------------------------------------------------------------------
// Kernel 4: output GEMM: ctx[4096x1024] f16 @ Wo[1024x1024] (f32->f16) + bo -> f32
// ---------------------------------------------------------------------------
__global__ __launch_bounds__(256) void k_out(const _Float16* __restrict__ ctx,
                                             const float* __restrict__ W,
                                             const float* __restrict__ bias,
                                             float* __restrict__ out) {
  __shared__ __align__(16) char smem[(128 * LDK + 64 * LDK) * 2];
  _Float16* ldsA  = (_Float16*)smem;
  _Float16* ldsBt = (_Float16*)(smem + 128 * LDK * 2);
  const int tid = threadIdx.x, wv = tid >> 5, l = tid & 31, lo = l & 15, hi = l >> 4;
  const int n0 = blockIdx.x * 64, m0 = blockIdx.y * 128;

  v8f acc[4];
  for (int f = 0; f < 4; ++f) for (int i = 0; i < 8; ++i) acc[f][i] = 0.f;

  for (int k0 = 0; k0 < H_; k0 += 32) {
    __syncthreads();
#pragma unroll
    for (int it = 0; it < 2; ++it) {
      const int i = tid + it * 256;
      const int r = i >> 2, c = (i & 3) * 8;
      async_copy16(&ldsA[r * LDK + c], &ctx[(size_t)(m0 + r) * H_ + k0 + c]);
    }
#pragma unroll
    for (int it = 0; it < 2; ++it) {
      const int i = tid + it * 256;
      const int kk = i >> 4, c = (i & 15) * 4;
      float4 w4 = *(const float4*)&W[(size_t)(k0 + kk) * H_ + n0 + c];
      if (k0 + 32 < H_)
        __builtin_prefetch(&W[(size_t)(k0 + 32 + kk) * H_ + n0 + c], 0, 1);
      ldsBt[(c + 0) * LDK + kk] = (_Float16)w4.x;
      ldsBt[(c + 1) * LDK + kk] = (_Float16)w4.y;
      ldsBt[(c + 2) * LDK + kk] = (_Float16)w4.z;
      ldsBt[(c + 3) * LDK + kk] = (_Float16)w4.w;
    }
    async_wait0();
    __syncthreads();
    V16 A;
    const _Float16* ap = &ldsA[(wv * 16 + lo) * LDK + hi * 8];
    A.h[0] = *(const v8h*)ap; A.h[1] = *(const v8h*)(ap + 16);
    V16 Bf[4];
#pragma unroll
    for (int f = 0; f < 4; ++f) {
      const _Float16* bp = &ldsBt[(f * 16 + lo) * LDK + hi * 8];
      Bf[f].h[0] = *(const v8h*)bp; Bf[f].h[1] = *(const v8h*)(bp + 16);
    }
#pragma unroll
    for (int f = 0; f < 4; ++f) acc[f] = wmma_f16(A.v, Bf[f].v, acc[f]);
  }
  for (int f = 0; f < 4; ++f) {
    const float bb = bias[n0 + f * 16 + lo];
    for (int r = 0; r < 8; ++r) {
      const int m = m0 + wv * 16 + hi * 8 + r;
      out[(size_t)m * H_ + n0 + f * 16 + lo] = acc[f][r] + bb;
    }
  }
}

// ---------------------------------------------------------------------------
extern "C" void kernel_launch(void* const* d_in, const int* in_sizes, int n_in,
                              void* d_out, int out_size, void* d_ws, size_t ws_size,
                              hipStream_t stream) {
  (void)in_sizes; (void)n_in; (void)out_size; (void)ws_size;
  const float* x     = (const float*)d_in[0];
  const int*   imask = (const int*)d_in[1];
  const float* nw    = (const float*)d_in[2];
  const float* nb    = (const float*)d_in[3];
  const float* Wqkv  = (const float*)d_in[4];
  const float* bqkv  = (const float*)d_in[5];
  const float* Wo    = (const float*)d_in[6];
  const float* bo    = (const float*)d_in[7];
  float* out = (float*)d_out;

  char* ws = (char*)d_ws;
  _Float16* XN = (_Float16*)(ws);                          // 8 MiB: xn f16
  _Float16* Qh = (_Float16*)(ws + (size_t)8  * 1048576);   // 8 MiB: Q [BH][S][HD]
  _Float16* Kh = (_Float16*)(ws + (size_t)16 * 1048576);   // 8 MiB: K [BH][S][HD]
  _Float16* Vt = (_Float16*)(ws + (size_t)24 * 1048576);   // 8 MiB: V [BH][HD][S]
  _Float16* CT = (_Float16*)(ws + (size_t)32 * 1048576);   // 8 MiB: ctx f16

  k_ln<<<M_, 256, 0, stream>>>(x, nw, nb, XN);

  dim3 g2(N3_ / 64, M_ / 128);           // 48 x 32
  k_qkv<<<g2, 256, 0, stream>>>(XN, Wqkv, bqkv, Qh, Kh, Vt);

  k_attn<<<(B_ * NH_ * (S_ / 16)) / 4, 128, 0, stream>>>(Qh, Kh, Vt, imask, CT);

  dim3 g4(H_ / 64, M_ / 128);            // 16 x 32
  k_out<<<g4, 256, 0, stream>>>(CT, Wo, bo, out);
}
